// HYV3MoEFused_90099823935489
// MI455X (gfx1250) — matrix operand order repaired and checked by reading, not compiled, not run on hardware
//
#include <hip/hip_runtime.h>

// ---------------- problem constants ----------------
constexpr int T_  = 1024;   // tokens
constexpr int H_  = 1024;   // hidden
constexpr int E_  = 16;     // routed experts
constexpr int I_  = 512;    // routed intermediate
constexpr int ISH = 1024;   // shared intermediate (I*NS)
constexpr float ROUTER_SCALE = 1.0f;

constexpr int TM = 128;     // M tile (tokens per block)

// ---------------- vector types ----------------
typedef __attribute__((ext_vector_type(16))) __bf16          v16bf;
typedef __attribute__((ext_vector_type(2)))  __bf16          v2bf;
typedef __attribute__((ext_vector_type(8)))  float           f32x8;
typedef __attribute__((ext_vector_type(4)))  float           f32x4;
typedef __attribute__((ext_vector_type(4)))  unsigned int    u32x4;

union FragCast { u32x4 u[2]; v16bf v; };

// fast reciprocal (v_rcp_f32)
__device__ __forceinline__ float frcp(float x) {
#if __has_builtin(__builtin_amdgcn_rcpf)
  return __builtin_amdgcn_rcpf(x);
#else
  return 1.f / x;
#endif
}

// fp32 -> bf16 (RNE-ish)
__device__ __forceinline__ unsigned short f2bf(float f) {
  unsigned int u = __builtin_bit_cast(unsigned int, f);
  unsigned int r = u + 0x7FFFu + ((u >> 16) & 1u);
  return (unsigned short)(r >> 16);
}

// pack two fp32 -> one dword of 2 bf16 (v_cvt_pk_bf16_f32 when available)
__device__ __forceinline__ unsigned cvt2bf(float a, float b) {
#if __has_builtin(__builtin_amdgcn_cvt_pk_bf16_f32)
  v2bf r = __builtin_amdgcn_cvt_pk_bf16_f32(a, b);
  return __builtin_bit_cast(unsigned, r);
#else
  return (unsigned)f2bf(a) | ((unsigned)f2bf(b) << 16);
#endif
}

// async copy of 16 bytes global -> LDS (gfx1250), tracked by ASYNCcnt
__device__ __forceinline__ void async_g2l_b128(void* lds, const void* gsrc) {
  unsigned lds_off = (unsigned)(unsigned long long)lds;   // low 32b = LDS offset
  asm volatile("global_load_async_to_lds_b128 %0, %1, off"
               :: "v"(lds_off), "v"(gsrc) : "memory");
}
__device__ __forceinline__ void wait_async0() {
  asm volatile("s_wait_asynccnt 0x0" ::: "memory");
}

// A-fragment (16x32 bf16): lane holds row M=lane&15, K {kb..kb+7} in v0-3 and
// {kb+16..kb+23} in v4-7 with kb = (lane>>4)*8.  Tile row-major [16][32].
__device__ __forceinline__ v16bf load_fragA(const unsigned short* base, int rowOff, int lane) {
  int r  = rowOff + (lane & 15);
  int kb = (lane >> 4) * 8;
  const unsigned short* p = base + r * 32 + kb;
  FragCast fc;
  fc.u[0] = *(const u32x4*)(p);
  fc.u[1] = *(const u32x4*)(p + 16);
  return fc.v;
}

// B-fragment (32x16 bf16): lane holds column N=lane&15, K {kb..kb+7} and
// {kb+16..kb+23}, kb = (lane>>4)*8.  Tile stored as [N][32] row-major.
__device__ __forceinline__ v16bf load_fragB(const unsigned short* base, int rowOff, int lane) {
  int r  = rowOff + (lane & 15);
  int kb = (lane >> 4) * 8;
  const unsigned short* p = base + r * 32 + kb;
  FragCast fc;
  fc.u[0] = *(const u32x4*)(p);
  fc.u[1] = *(const u32x4*)(p + 16);
  return fc.v;
}

__device__ __forceinline__ f32x8 zero8() {
  f32x8 z = {0.f,0.f,0.f,0.f,0.f,0.f,0.f,0.f};
  return z;
}

// ---------------- kernel 1: fp32 -> bf16 copy ----------------
__global__ void k_cvt(const float* __restrict__ src, unsigned short* __restrict__ dst, int n) {
  int i = (blockIdx.x * blockDim.x + threadIdx.x) * 8;
  if (i + 7 < n) {
    f32x4 f0 = *(const f32x4*)(src + i);
    f32x4 f1 = *(const f32x4*)(src + i + 4);
    u32x4 o;
    o.x = cvt2bf(f0.x, f0.y); o.y = cvt2bf(f0.z, f0.w);
    o.z = cvt2bf(f1.x, f1.y); o.w = cvt2bf(f1.z, f1.w);
    *(u32x4*)(dst + i) = o;
  }
}

// ---------------- kernel 2: router (one wave / token) ----------------
__global__ void k_router(const float* __restrict__ x, const float* __restrict__ gw,
                         const float* __restrict__ bias,
                         int* __restrict__ top_idx, float* __restrict__ top_w) {
  int token = blockIdx.x * 8 + (threadIdx.x >> 5);
  int lane  = threadIdx.x & 31;
  int e     = lane & 15;
  int half  = lane >> 4;
  const f32x4* xr = (const f32x4*)(x  + (size_t)token * H_) + half * (H_ / 8);
  const f32x4* wr = (const f32x4*)(gw + (size_t)e     * H_) + half * (H_ / 8);
  float acc = 0.f;
  for (int k = 0; k < H_ / 8; ++k) {
    f32x4 a = xr[k], b = wr[k];
    acc += a.x * b.x + a.y * b.y + a.z * b.z + a.w * b.w;
  }
  acc += __shfl_down(acc, 16, 32);           // lanes 0..15 hold full logits
  float score  = frcp(1.f + __expf(-acc));   // sigmoid
  float biased = score + bias[e];
  int   b0 = 0, b1 = 0;
  float v0 = -1e30f, v1 = -1e30f, s0 = 0.f, s1 = 0.f;
  for (int i = 0; i < 16; ++i) {
    float bv = __shfl(biased, i, 32);
    float sv = __shfl(score,  i, 32);
    if (bv > v0) { v1 = v0; b1 = b0; s1 = s0; v0 = bv; b0 = i; s0 = sv; }
    else if (bv > v1) { v1 = bv; b1 = i; s1 = sv; }
  }
  float inv = ROUTER_SCALE * frcp(s0 + s1);
  if (lane == 0) {
    top_idx[2 * token]     = b0;
    top_idx[2 * token + 1] = b1;
    top_w [2 * token]      = s0 * inv;
    top_w [2 * token + 1]  = s1 * inv;
  }
}

// ---------------- kernel 3: per-expert token-list build (deterministic) ----
__global__ void k_build(const int* __restrict__ top_idx, const float* __restrict__ top_w,
                        int* __restrict__ tok_list, float* __restrict__ wt_list,
                        int* __restrict__ counts) {
  int e = blockIdx.x;
  int t = threadIdx.x;          // blockDim.x == T_
  int i0 = top_idx[2 * t], i1 = top_idx[2 * t + 1];
  bool pres = (i0 == e) || (i1 == e);
  float w   = (i0 == e) ? top_w[2 * t] : top_w[2 * t + 1];
  unsigned mask = (unsigned)__ballot(pres ? 1 : 0);   // wave32: low 32 bits
  int lane = t & 31, wv = t >> 5;
  __shared__ int wc[32];
  if (lane == 0) wc[wv] = __popc(mask);
  __syncthreads();
  if (t == 0) {
    int s = 0;
    for (int i = 0; i < 32; ++i) { int c = wc[i]; wc[i] = s; s += c; }
    counts[e] = s;
  }
  __syncthreads();
  if (pres) {
    int slot = wc[wv] + __popc(mask & ((1u << lane) - 1u));
    tok_list[e * T_ + slot] = t;
    wt_list [e * T_ + slot] = w;
  }
}

// ---------------- kernel 4: gate_up GEMM + fused SwiGLU -------------------
// Block: 256 thr = 8 waves.  Tile: M=128 tokens x 64 paired (g,u) columns.
// Double-buffered LDS; A tile staged via async global->LDS; B tile converted
// fp32->bf16 in flight (weights are touched exactly once -> inline cvt is the
// bandwidth-optimal path).
__global__ __launch_bounds__(256)
void k_mlp_up(const unsigned short* __restrict__ xbf,
              const float* __restrict__ W, long long eStrideW, int Ipart,
              const int* __restrict__ toklist, const int* __restrict__ counts,
              unsigned short* __restrict__ hout, long long hExpertStride) {
  int e   = blockIdx.z;
  int cnt = counts ? counts[e] : T_;
  int m0  = blockIdx.y * TM;
  if (m0 >= cnt) return;
  int n0  = blockIdx.x * 64;
  const float* We = W + (long long)e * eStrideW;

  __shared__ __align__(16) unsigned short As[2][TM * 32];
  __shared__ __align__(16) unsigned short Bs[2][TM * 32];
  __shared__ int toks[TM];

  int tid = threadIdx.x;
  if (tid < TM) {
    int r = m0 + tid;
    int rc = (r < cnt) ? r : (cnt - 1);
    toks[tid] = toklist ? toklist[e * T_ + rc] : rc;
  }
  int lane = tid & 31, wv = tid >> 5;
  int wm = wv & 3, wn = wv >> 2;    // wave tile: rows 32*wm, paired cols 32*wn

  f32x8 cg[2][2], cu[2][2];
#pragma unroll
  for (int i = 0; i < 2; ++i)
#pragma unroll
    for (int j = 0; j < 2; ++j) { cg[i][j] = zero8(); cu[i][j] = zero8(); }

  __syncthreads();   // toks visible

  auto stage = [&](int k0, int buf) {
    // A tile (bf16 gathered token rows): async copy, 16B per lane per pass
#pragma unroll
    for (int p = 0; p < 2; ++p) {
      int idx = tid + p * 256;
      int r = idx >> 2, c8 = (idx & 3) * 8;
      const unsigned short* src = xbf + (long long)toks[r] * H_ + k0 + c8;
      async_g2l_b128(&As[buf][r * 32 + c8], src);
    }
    // B tile (weight rows fp32 -> bf16): rows 0..63 = g, 64..127 = u
#pragma unroll
    for (int p = 0; p < 2; ++p) {
      int idx = tid + p * 256;
      int r = idx >> 2, c8 = (idx & 3) * 8;
      int grow = (r < 64) ? (n0 + r) : (Ipart + n0 + (r - 64));
      const float* wsrc = We + (long long)grow * H_ + k0 + c8;
      __builtin_prefetch(wsrc + 32, 0, 0);   // next K-chunk cacheline
      f32x4 f0 = ((const f32x4*)wsrc)[0];
      f32x4 f1 = ((const f32x4*)wsrc)[1];
      u32x4 o;
      o.x = cvt2bf(f0.x, f0.y); o.y = cvt2bf(f0.z, f0.w);
      o.z = cvt2bf(f1.x, f1.y); o.w = cvt2bf(f1.z, f1.w);
      *(u32x4*)&Bs[buf][r * 32 + c8] = o;
    }
  };

  auto compute = [&](int buf) {
    v16bf af[2], bg[2], bu[2];
#pragma unroll
    for (int i = 0; i < 2; ++i) af[i] = load_fragA(As[buf], 32 * wm + 16 * i, lane);
#pragma unroll
    for (int j = 0; j < 2; ++j) {
      bg[j] = load_fragB(Bs[buf],      32 * wn + 16 * j, lane);
      bu[j] = load_fragB(Bs[buf], 64 + 32 * wn + 16 * j, lane);
    }
#pragma unroll
    for (int i = 0; i < 2; ++i)
#pragma unroll
      for (int j = 0; j < 2; ++j) {
        cg[i][j] = __builtin_amdgcn_wmma_f32_16x16x32_bf16(
            false, af[i], false, bg[j], (short)0, cg[i][j], false, false);
        cu[i][j] = __builtin_amdgcn_wmma_f32_16x16x32_bf16(
            false, af[i], false, bu[j], (short)0, cu[i][j], false, false);
      }
  };

  stage(0, 0);
  int cur = 0;
  for (int k0 = 32; k0 < H_; k0 += 32) {
    wait_async0();          // own async A writes done
    __syncthreads();        // everyone's stage(cur) visible
    stage(k0, cur ^ 1);     // overlap next stage with compute
    compute(cur);
    cur ^= 1;
  }
  wait_async0();
  __syncthreads();
  compute(cur);

  // fused SwiGLU epilogue -> bf16 activation scratch
  unsigned short* hE = hout + (long long)e * hExpertStride;
#pragma unroll
  for (int i = 0; i < 2; ++i)
#pragma unroll
    for (int j = 0; j < 2; ++j)
#pragma unroll
      for (int v = 0; v < 8; ++v) {
        int m = m0 + 32 * wm + 16 * i + v + 8 * (lane >> 4);
        if (m < cnt) {
          int n = n0 + 32 * wn + 16 * j + (lane & 15);
          float g = cg[i][j][v], u = cu[i][j][v];
          float h = g * frcp(1.f + __expf(-g)) * u;   // silu(g)*u
          hE[(long long)m * Ipart + n] = f2bf(h);
        }
      }
}

// ---------------- kernel 5: down GEMM + combine ---------------------------
// Block: 256 thr = 8 waves.  Tile: M=128 x N=128 (H cols), K = Kdim.
// wt_list==nullptr -> plain store (shared expert, initializes out);
// else atomicAdd of weight-scaled rows into out (routed scatter).
__global__ __launch_bounds__(256)
void k_mlp_down(const unsigned short* __restrict__ hin, long long hExpertStride, int Kdim,
                const float* __restrict__ W, long long eStrideW,
                const int* __restrict__ toklist, const int* __restrict__ counts,
                const float* __restrict__ wt_list,
                float* __restrict__ out) {
  int e   = blockIdx.z;
  int cnt = counts ? counts[e] : T_;
  int m0  = blockIdx.y * TM;
  if (m0 >= cnt) return;
  int n0  = blockIdx.x * TM;
  const float*          We = W   + (long long)e * eStrideW;
  const unsigned short* hE = hin + (long long)e * hExpertStride;

  __shared__ __align__(16) unsigned short As[2][TM * 32];
  __shared__ __align__(16) unsigned short Bs[2][TM * 32];

  int tid = threadIdx.x;
  int lane = tid & 31, wv = tid >> 5;
  int wm = wv & 3, wn = wv >> 2;    // rows 32*wm, cols 64*wn

  f32x8 acc[2][4];
#pragma unroll
  for (int i = 0; i < 2; ++i)
#pragma unroll
    for (int j = 0; j < 4; ++j) acc[i][j] = zero8();

  auto stage = [&](int k0, int buf) {
#pragma unroll
    for (int p = 0; p < 2; ++p) {
      int idx = tid + p * 256;
      int r = idx >> 2, c8 = (idx & 3) * 8;
      const unsigned short* src = hE + (long long)(m0 + r) * Kdim + k0 + c8;
      async_g2l_b128(&As[buf][r * 32 + c8], src);
    }
#pragma unroll
    for (int p = 0; p < 2; ++p) {
      int idx = tid + p * 256;
      int r = idx >> 2, c8 = (idx & 3) * 8;
      const float* wsrc = We + (long long)(n0 + r) * Kdim + k0 + c8;
      __builtin_prefetch(wsrc + 32, 0, 0);
      f32x4 f0 = ((const f32x4*)wsrc)[0];
      f32x4 f1 = ((const f32x4*)wsrc)[1];
      u32x4 o;
      o.x = cvt2bf(f0.x, f0.y); o.y = cvt2bf(f0.z, f0.w);
      o.z = cvt2bf(f1.x, f1.y); o.w = cvt2bf(f1.z, f1.w);
      *(u32x4*)&Bs[buf][r * 32 + c8] = o;
    }
  };

  auto compute = [&](int buf) {
    v16bf af[2], bb[4];
#pragma unroll
    for (int i = 0; i < 2; ++i) af[i] = load_fragA(As[buf], 32 * wm + 16 * i, lane);
#pragma unroll
    for (int j = 0; j < 4; ++j) bb[j] = load_fragB(Bs[buf], 64 * wn + 16 * j, lane);
#pragma unroll
    for (int i = 0; i < 2; ++i)
#pragma unroll
      for (int j = 0; j < 4; ++j)
        acc[i][j] = __builtin_amdgcn_wmma_f32_16x16x32_bf16(
            false, af[i], false, bb[j], (short)0, acc[i][j], false, false);
  };

  stage(0, 0);
  int cur = 0;
  for (int k0 = 32; k0 < Kdim; k0 += 32) {
    wait_async0();
    __syncthreads();
    stage(k0, cur ^ 1);
    compute(cur);
    cur ^= 1;
  }
  wait_async0();
  __syncthreads();
  compute(cur);

#pragma unroll
  for (int i = 0; i < 2; ++i)
#pragma unroll
    for (int j = 0; j < 4; ++j)
#pragma unroll
      for (int v = 0; v < 8; ++v) {
        int m = m0 + 32 * wm + 16 * i + v + 8 * (lane >> 4);
        if (m < cnt) {
          int n = n0 + 64 * wn + 16 * j + (lane & 15);
          float y = acc[i][j][v];
          if (wt_list) {
            int t = toklist[e * T_ + m];
            atomicAdd(out + (long long)t * H_ + n, wt_list[e * T_ + m] * y);
          } else {
            out[(long long)m * H_ + n] = y;
          }
        }
      }
}

// ---------------- host launcher ----------------
extern "C" void kernel_launch(void* const* d_in, const int* in_sizes, int n_in,
                              void* d_out, int out_size, void* d_ws, size_t ws_size,
                              hipStream_t stream) {
  (void)in_sizes; (void)n_in; (void)out_size; (void)ws_size;
  const float* x     = (const float*)d_in[0];   // [T,H]
  const float* gw    = (const float*)d_in[1];   // [E,H]
  const float* ebias = (const float*)d_in[2];   // [E]
  const float* wgu   = (const float*)d_in[3];   // [E,2I,H]
  const float* wdn   = (const float*)d_in[4];   // [E,H,I]
  const float* sgu   = (const float*)d_in[5];   // [2*ISH,H]
  const float* sdn   = (const float*)d_in[6];   // [H,ISH]
  float* out = (float*)d_out;

  // workspace carve-up (~21.1 MB total)
  char* ws = (char*)d_ws;
  size_t off = 0;
  auto carve = [&](size_t bytes) -> void* {
    off = (off + 255) & ~(size_t)255;
    void* p = ws + off;
    off += bytes;
    return p;
  };
  unsigned short* xbf     = (unsigned short*)carve((size_t)T_ * H_ * 2);
  int*            top_idx = (int*)           carve((size_t)T_ * 2 * 4);
  float*          top_w   = (float*)         carve((size_t)T_ * 2 * 4);
  int*            counts  = (int*)           carve((size_t)E_ * 4);
  int*            tok_l   = (int*)           carve((size_t)E_ * T_ * 4);
  float*          wt_l    = (float*)         carve((size_t)E_ * T_ * 4);
  unsigned short* h_rout  = (unsigned short*)carve((size_t)E_ * T_ * I_ * 2);
  unsigned short* h_sh    = (unsigned short*)carve((size_t)T_ * ISH * 2);

  // 1) x -> bf16
  k_cvt<<<(T_ * H_) / (256 * 8), 256, 0, stream>>>(x, xbf, T_ * H_);
  // 2) router top-2
  k_router<<<T_ / 8, 256, 0, stream>>>(x, gw, ebias, top_idx, top_w);
  // 3) per-expert compaction
  k_build<<<E_, T_, 0, stream>>>(top_idx, top_w, tok_l, wt_l, counts);
  // 4) shared expert: gate_up + SwiGLU
  k_mlp_up<<<dim3(ISH / 64, T_ / TM, 1), 256, 0, stream>>>(
      xbf, sgu, 0LL, ISH, nullptr, nullptr, h_sh, 0LL);
  // 5) shared expert: down-proj (plain store, initializes out)
  k_mlp_down<<<dim3(H_ / TM, T_ / TM, 1), 256, 0, stream>>>(
      h_sh, 0LL, ISH, sdn, 0LL, nullptr, nullptr, nullptr, out);
  // 6) routed experts: gate_up + SwiGLU per expert token list
  k_mlp_up<<<dim3(I_ / 64, T_ / TM, E_), 256, 0, stream>>>(
      xbf, wgu, (long long)2 * I_ * H_, I_, tok_l, counts, h_rout, (long long)T_ * I_);
  // 7) routed experts: down-proj, weighted atomic scatter into out
  k_mlp_down<<<dim3(H_ / TM, T_ / TM, E_), 256, 0, stream>>>(
      h_rout, (long long)T_ * I_, I_, wdn, (long long)H_ * I_, tok_l, counts, wt_l, out);
}
